// SAE_62139586839260
// MI455X (gfx1250) — matrix operand order repaired and checked by reading, not compiled, not run on hardware
//
#include <hip/hip_runtime.h>
#include <hip/hip_bf16.h>
#include <stdint.h>

// ---------------------------------------------------------------------------
// SAE encoder: z = relu(x @ W + b); keep top-k per row, zero the rest.
// GEMM: bf16 WMMA (v_wmma_f32_16x16x32_bf16), 128x128x64 block tiles,
// double-buffered LDS, A tile staged with global_load_async_to_lds_b128.
// ---------------------------------------------------------------------------

typedef __attribute__((ext_vector_type(16))) __bf16 bf16x16;
typedef __attribute__((ext_vector_type(8)))  float  floatx8;

#define TILE_M 128
#define TILE_N 128
#define TILE_K 64
#define THREADS 256
#define A_STRIDE 72   // bf16 units (144 B rows: 16B-aligned, conflict-free)
#define B_STRIDE 36   // u32 units  (144 B rows: 16B-aligned, conflict-free)

union Frag {
    bf16x16 v;
    uint4   q[2];
};

__global__ __launch_bounds__(THREADS)
void sae_gemm_bias_relu(const __hip_bfloat16* __restrict__ X,
                        const __hip_bfloat16* __restrict__ W,
                        const __hip_bfloat16* __restrict__ Bias,
                        __hip_bfloat16* __restrict__ Z,
                        int N, int D, int F)
{
    __shared__ __align__(16) uint16_t As[2][TILE_M * A_STRIDE]; // row-major x tile
    __shared__ __align__(16) uint32_t Bs[2][TILE_N * B_STRIDE]; // W tile, transposed k-pairs

    const int tid   = threadIdx.x;
    const int lane  = tid & 31;
    const int wid   = tid >> 5;
    const int waveM = wid & 3;   // 4 wave rows -> 32 M each
    const int waveN = wid >> 2;  // 2 wave cols -> 64 N each
    // blockIdx.x = M (fast-moving): consecutive blocks share the same W panel,
    // and the whole x matrix (32 MB) stays resident in the 192 MB L2.
    const int m0 = blockIdx.x * TILE_M;
    const int n0 = blockIdx.y * TILE_N;

    const uint16_t* Xu = (const uint16_t*)X;
    const uint16_t* Wu = (const uint16_t*)W;

    floatx8 acc[2][4];
#pragma unroll
    for (int i = 0; i < 2; ++i)
#pragma unroll
        for (int j = 0; j < 4; ++j)
            acc[i][j] = (floatx8){0.f,0.f,0.f,0.f,0.f,0.f,0.f,0.f};

    // Per-lane bias for each of the 4 N-tiles this wave owns.
    float bias[4];
#pragma unroll
    for (int nt = 0; nt < 4; ++nt) {
        int gn = n0 + waveN * 64 + nt * 16 + (lane & 15);
        bias[nt] = __bfloat162float(Bias[gn]);
    }

    // ---- staging helpers -------------------------------------------------
    // A tile: 128x64 bf16 copied verbatim -> async DMA to LDS (ASYNCcnt).
    auto stageA = [&](int buf, int kt) {
#pragma unroll
        for (int i = 0; i < 4; ++i) {
            int s   = i * THREADS + tid;   // 16-byte segment id 0..1023
            int row = s >> 3;              // 0..127
            int cs  = s & 7;               // 0..7
            const void* g = (const void*)(Xu + (size_t)(m0 + row) * D + kt + cs * 8);
            uint32_t l = (uint32_t)(size_t)&As[buf][row * A_STRIDE + cs * 8];
            asm volatile("global_load_async_to_lds_b128 %0, %1, off"
                         :: "v"(l), "v"(g) : "memory");
        }
    };
    // B tile: 64(k) x 128(n); load coalesced row-pairs into registers...
    auto loadB = [&](int kt, uint4 r[2][2]) {
#pragma unroll
        for (int i = 0; i < 2; ++i) {
            int p  = i * THREADS + tid;    // 0..511
            int rp = p >> 4;               // k-pair 0..31
            int cs = p & 15;               // 8-col segment 0..15
            const uint16_t* g0 = Wu + (size_t)(kt + 2 * rp) * F + n0 + cs * 8;
            r[i][0] = *(const uint4*)g0;
            r[i][1] = *(const uint4*)(g0 + F);
        }
    };
    // ...then pack (k,k+1) pairs and scatter transposed into LDS.
    auto storeB = [&](int buf, const uint4 r[2][2]) {
#pragma unroll
        for (int i = 0; i < 2; ++i) {
            int p  = i * THREADS + tid;
            int rp = p >> 4;
            int cs = p & 15;
            const uint16_t* h0 = (const uint16_t*)&r[i][0];
            const uint16_t* h1 = (const uint16_t*)&r[i][1];
#pragma unroll
            for (int j = 0; j < 8; ++j)
                Bs[buf][(cs * 8 + j) * B_STRIDE + rp] =
                    (uint32_t)h0[j] | ((uint32_t)h1[j] << 16);
        }
    };

    auto computeTile = [&](int buf) {
#pragma unroll
        for (int k0 = 0; k0 < TILE_K; k0 += 32) {
            Frag fa[2], fb[4];
#pragma unroll
            for (int mt = 0; mt < 2; ++mt) {
                // ISA A layout: lanes 0-15 -> K[k0..+7],K[k0+16..+23]; lanes 16-31 -> +8
                int rowA = waveM * 32 + mt * 16 + (lane & 15);
                int base = rowA * A_STRIDE + k0 + (lane >> 4) * 8;
                fa[mt].q[0] = *(const uint4*)&As[buf][base];
                fa[mt].q[1] = *(const uint4*)&As[buf][base + 16];
            }
#pragma unroll
            for (int nt = 0; nt < 4; ++nt) {
                // B layout (matches A): lanes 0-15 -> K[k0..+7],K[k0+16..+23]; lanes 16-31 -> +8
                int col  = waveN * 64 + nt * 16 + (lane & 15);
                int base = col * B_STRIDE + (k0 >> 1) + (lane >> 4) * 4;
                fb[nt].q[0] = *(const uint4*)&Bs[buf][base];
                fb[nt].q[1] = *(const uint4*)&Bs[buf][base + 8];
            }
#pragma unroll
            for (int mt = 0; mt < 2; ++mt)
#pragma unroll
                for (int nt = 0; nt < 4; ++nt)
                    acc[mt][nt] = __builtin_amdgcn_wmma_f32_16x16x32_bf16(
                        false, fa[mt].v, false, fb[nt].v,
                        (short)0, acc[mt][nt], false, false);
        }
    };

    // ---- software-pipelined main loop (double buffer, 1 barrier/iter) ----
    uint4 rB[2][2];
    const int nTiles = D / TILE_K;

    stageA(0, 0);
    loadB(0, rB);
    storeB(0, rB);

    for (int t = 0; t < nTiles; ++t) {
        const int cur = t & 1;
        asm volatile("s_wait_asynccnt 0x0" ::: "memory"); // my async A writes done
        __syncthreads();                                  // everyone's buf[cur] ready
        const bool more = (t + 1) < nTiles;
        if (more) {
            stageA(cur ^ 1, (t + 1) * TILE_K);            // DMA next A tile
            loadB((t + 1) * TILE_K, rB);                  // next B rows in flight
        }
        computeTile(cur);                                 // 16 WMMAs hide the loads
        if (more)
            storeB(cur ^ 1, rB);                          // pack/scatter next B tile
    }

    // ---- Epilogue: bias + relu, f32 -> bf16, store ----
#pragma unroll
    for (int mt = 0; mt < 2; ++mt) {
#pragma unroll
        for (int nt = 0; nt < 4; ++nt) {
            int gn = n0 + waveN * 64 + nt * 16 + (lane & 15);
#pragma unroll
            for (int v = 0; v < 8; ++v) {
                int gm = m0 + waveM * 32 + mt * 16 + v + ((lane >> 4) * 8);
                float val = acc[mt][nt][v] + bias[nt];
                val = val > 0.0f ? val : 0.0f;
                Z[(size_t)gm * F + gn] = __float2bfloat16(val);
            }
        }
    }
}

// ---------------------------------------------------------------------------
// Per-row top-k via 2-level radix select on bf16 bit patterns (values >= 0
// after ReLU so u16 pattern ordering == float ordering). Deterministic tie
// handling via a prefix scan of per-thread tie counts.
// ---------------------------------------------------------------------------
__global__ __launch_bounds__(1024)
void sae_topk(__hip_bfloat16* __restrict__ Z, const int* __restrict__ kp, int F)
{
    __shared__ uint32_t hist[256];
    __shared__ uint32_t tiePre[1024];
    __shared__ uint32_t sBin, sRank, sThresh, sTieKeep;

    const int tid = threadIdx.x;
    const uint32_t k = (uint32_t)(*kp);
    uint16_t* row  = (uint16_t*)Z + (size_t)blockIdx.x * F;
    uint4*    row4 = (uint4*)row;
    const int nVec = F >> 3;

    // ---- Pass 1: histogram of high byte ----
    for (int i = tid; i < 256; i += 1024) hist[i] = 0;
    __syncthreads();
    for (int i = tid; i < nVec; i += 1024) {
        uint4 v = row4[i];
        const uint16_t* h = (const uint16_t*)&v;
#pragma unroll
        for (int j = 0; j < 8; ++j) atomicAdd(&hist[h[j] >> 8], 1u);
    }
    __syncthreads();
    if (tid == 0) {
        uint32_t cum = 0; uint32_t b = 0;
        for (int i = 255; i >= 0; --i) {
            if (cum + hist[i] >= k) { b = (uint32_t)i; break; }
            cum += hist[i];
        }
        sBin = b; sRank = k - cum;   // rank within selected high-byte bin, >=1
    }
    __syncthreads();
    const uint32_t binSel = sBin;
    const uint32_t rank   = sRank;

    // ---- Pass 2: histogram of low byte within selected bin ----
    for (int i = tid; i < 256; i += 1024) hist[i] = 0;
    __syncthreads();
    for (int i = tid; i < nVec; i += 1024) {
        uint4 v = row4[i];
        const uint16_t* h = (const uint16_t*)&v;
#pragma unroll
        for (int j = 0; j < 8; ++j)
            if ((uint32_t)(h[j] >> 8) == binSel) atomicAdd(&hist[h[j] & 0xFF], 1u);
    }
    __syncthreads();
    if (tid == 0) {
        uint32_t cum = 0; uint32_t L = 0;
        for (int i = 255; i >= 0; --i) {
            if (cum + hist[i] >= rank) { L = (uint32_t)i; break; }
            cum += hist[i];
        }
        sThresh  = (binSel << 8) | L;   // exact k-th largest bit pattern
        sTieKeep = rank - cum;          // how many ==thresh survive
    }
    __syncthreads();
    const uint32_t thresh  = sThresh;
    const uint32_t tieKeep = sTieKeep;

    // ---- Pass 3: deterministic prefix of per-thread tie counts ----
    uint32_t myTies = 0;
    for (int i = tid; i < nVec; i += 1024) {
        uint4 v = row4[i];
        const uint16_t* h = (const uint16_t*)&v;
#pragma unroll
        for (int j = 0; j < 8; ++j) myTies += (uint32_t)(h[j] == (uint16_t)thresh);
    }
    tiePre[tid] = myTies;
    __syncthreads();
    if (tid == 0) {
        uint32_t run = 0;
        for (int i = 0; i < 1024; ++i) { uint32_t c = tiePre[i]; tiePre[i] = run; run += c; }
    }
    __syncthreads();
    int32_t keepTies = (int32_t)tieKeep - (int32_t)tiePre[tid];

    // ---- Pass 4: zero everything below threshold (and surplus ties) ----
    for (int i = tid; i < nVec; i += 1024) {
        uint4 v = row4[i];
        uint16_t* h = (uint16_t*)&v;
#pragma unroll
        for (int j = 0; j < 8; ++j) {
            uint16_t u = h[j];
            if ((uint32_t)u > thresh) continue;               // strictly above: keep
            if (u == (uint16_t)thresh && keepTies > 0) {      // tie: keep in fixed order
                --keepTies;
                continue;
            }
            h[j] = 0;
        }
        row4[i] = v;
    }
}

// ---------------------------------------------------------------------------
extern "C" void kernel_launch(void* const* d_in, const int* in_sizes, int n_in,
                              void* d_out, int out_size, void* d_ws, size_t ws_size,
                              hipStream_t stream)
{
    const __hip_bfloat16* x  = (const __hip_bfloat16*)d_in[0];
    const __hip_bfloat16* W  = (const __hip_bfloat16*)d_in[1];
    const __hip_bfloat16* be = (const __hip_bfloat16*)d_in[2];
    const int*            kp = (const int*)d_in[3];
    __hip_bfloat16* out = (__hip_bfloat16*)d_out;

    const int F = in_sizes[2];          // 65536
    const int D = in_sizes[1] / F;      // 4096
    const int N = in_sizes[0] / D;      // 4096

    dim3 grid(N / TILE_M, F / TILE_N);  // x = M (fast) for W-panel reuse in L2
    sae_gemm_bias_relu<<<grid, THREADS, 0, stream>>>(x, W, be, out, N, D, F);
    sae_topk<<<N, 1024, 0, stream>>>(out, kp, F);
}
